// LTC_76295799046671
// MI455X (gfx1250) — matrix operand order repaired and checked by reading, not compiled, not run on hardware
//
#include <hip/hip_runtime.h>
#include <hip/hip_bf16.h>
#include <math.h>

typedef __attribute__((ext_vector_type(16))) __bf16 v16bf;
typedef __attribute__((ext_vector_type(8)))  __bf16 v8bf;
typedef __attribute__((ext_vector_type(8)))  float  v8f;

union U16bf { v16bf v; v8bf h[2]; };

#define LTC_BATCH  128
#define LTC_SEQ    256
#define LTC_IN     512
#define LTC_HID    1024
#define LTC_OUTW   256
#define LTC_NBLK   64
#define PAD_WHH    (LTC_HID + 8)   // row stride (bf16 elems) -> 2064B, 16B aligned, conflict-free
#define PAD_WIN    (LTC_IN + 8)    // 1040B

// ---------------- fp32 -> bf16 conversion (grid-stride) ----------------
__global__ void ltc_cvt_bf16(const float* __restrict__ in, __bf16* __restrict__ out, int n) {
    int i = blockIdx.x * blockDim.x + threadIdx.x;
    int stride = gridDim.x * blockDim.x;
    for (; i < n; i += stride) out[i] = (__bf16)in[i];
}

// ---------------- per-channel coefficients + state init ----------------
__global__ void ltc_prep(const float* __restrict__ log_tau, const float* __restrict__ gleak,
                         const float* __restrict__ cm_t, const float* __restrict__ ts,
                         float4* __restrict__ coef, __bf16* __restrict__ hb0,
                         __bf16* __restrict__ hb1, unsigned int* __restrict__ bar) {
    int gid = blockIdx.x * blockDim.x + threadIdx.x;   // grid covers 128*1024 = 131072
    if (gid < LTC_HID) {
        float lt = log_tau[gid];
        float sp = (lt > 20.f) ? lt : log1pf(__expf(lt));   // softplus
        float tau = sp + 0.001f;
        float denom = cm_t[gid] + gleak[gid];
        float dt0 = ts[0];                   // uniform spacing -> 2 full MAX_DT + remainder
        float rem = dt0 - 2.0f * 0.02f;
        float a1 = 1.f - 0.02f / tau, b1 = 0.02f / denom;
        float a2, b2;
        if (rem > 1e-8f) { a2 = 1.f - rem / tau; b2 = rem / denom; }
        else             { a2 = 1.f;             b2 = 0.f; }       // identity substep
        coef[gid] = make_float4(a1, b1, a2, b2);
    }
    if (gid < LTC_BATCH * LTC_HID) { hb0[gid] = (__bf16)0.0f; hb1[gid] = (__bf16)0.0f; }
    if (gid == 0) *bar = 0u;
}

// ---------------- persistent recurrent kernel ----------------
// grid = 64 blocks x 256 threads. Block b owns hidden columns [16b, 16b+16).
__global__ __launch_bounds__(256, 1) void ltc_persistent(
    const __bf16* __restrict__ Xb,      // [B, S, IN] bf16
    const __bf16* __restrict__ Winb,    // [HID, IN]  bf16
    const __bf16* __restrict__ Whhb,    // [HID, HID] bf16
    __bf16* __restrict__ hb0,           // [B, HID] bf16 (double buffer A)
    __bf16* __restrict__ hb1,           // [B, HID] bf16 (double buffer B)
    __bf16* __restrict__ Hhist,         // [S*B, HID] bf16 per-timestep state
    const float* __restrict__ b_in, const float* __restrict__ b_hh,
    const float* __restrict__ mu, const float* __restrict__ sigma,
    const float4* __restrict__ coef,    // a1,b1,a2,b2 per hidden channel
    float* __restrict__ hfinal,         // [B, HID] fp32 (tail of d_out)
    unsigned int* __restrict__ bar)
{
    __shared__ __bf16 ldsW[16 * PAD_WHH];   // W_hh strip: 16 cols x 1024 K
    __shared__ __bf16 ldsI[16 * PAD_WIN];   // W_in strip: 16 cols x 512 K

    const int tid  = threadIdx.x;
    const int blk  = blockIdx.x;
    const int wave = tid >> 5;
    const int lane = tid & 31;
    const int hi   = lane >> 4;     // half-wave
    const int lq   = lane & 15;

    // stage weight strips into LDS once; they persist for all 256 timesteps
    for (int i = tid; i < 16 * LTC_HID; i += 256) {
        int r = i >> 10, k = i & (LTC_HID - 1);
        ldsW[r * PAD_WHH + k] = Whhb[(size_t)(blk * 16 + r) * LTC_HID + k];
    }
    for (int i = tid; i < 16 * LTC_IN; i += 256) {
        int r = i >> 9, k = i & (LTC_IN - 1);
        ldsI[r * PAD_WIN + k] = Winb[(size_t)(blk * 16 + r) * LTC_IN + k];
    }
    __syncthreads();

    const int col   = blk * 16 + lq;       // this lane's hidden channel (C/D col)
    const int arow  = wave * 16 + lq;      // A-operand batch row for this lane
    const int drow0 = wave * 16 + hi * 8;  // C/D rows drow0..drow0+7

    const float bin_c = b_in[col], bhh_c = b_hh[col];
    const float mu_c = mu[col], sg_c = sigma[col];
    const float4 cf = coef[col];

    float h[8];
    #pragma unroll
    for (int r = 0; r < 8; ++r) h[r] = 0.f;

    __bf16* cur = hb0;   // read buffer (zero-initialized by ltc_prep)
    __bf16* nxt = hb1;   // write buffer
    unsigned int stepno = 0;

    for (int t = 0; t < LTC_SEQ; ++t) {
        // ---- ie = x_t @ W_in^T (+ b_in), kept in registers ----
        v8f ie = {};
        const __bf16* xrow = Xb + ((size_t)arow * LTC_SEQ + t) * LTC_IN;
        #pragma unroll 4
        for (int kb = 0; kb < LTC_IN; kb += 32) {
            U16bf a, b;
            a.h[0] = *(const v8bf*)(xrow + kb + hi * 8);
            a.h[1] = *(const v8bf*)(xrow + kb + hi * 8 + 16);
            b.h[0] = *(const v8bf*)(&ldsI[lq * PAD_WIN + kb + hi * 8]);
            b.h[1] = *(const v8bf*)(&ldsI[lq * PAD_WIN + kb + hi * 8 + 16]);
            ie = __builtin_amdgcn_wmma_f32_16x16x32_bf16(false, a.v, false, b.v,
                                                         (short)0, ie, false, false);
        }
        #pragma unroll
        for (int r = 0; r < 8; ++r) ie[r] += bin_c;

        // ---- 3 substeps: dt = 0.02, 0.02, rem ----
        #pragma unroll 1
        for (int s = 0; s < 3; ++s) {
            v8f acc = ie;
            const __bf16* hrow = cur + (size_t)arow * LTC_HID;
            #pragma unroll 4
            for (int kb = 0; kb < LTC_HID; kb += 32) {
                U16bf a, b;
                a.h[0] = *(const v8bf*)(hrow + kb + hi * 8);
                a.h[1] = *(const v8bf*)(hrow + kb + hi * 8 + 16);
                b.h[0] = *(const v8bf*)(&ldsW[lq * PAD_WHH + kb + hi * 8]);
                b.h[1] = *(const v8bf*)(&ldsW[lq * PAD_WHH + kb + hi * 8 + 16]);
                acc = __builtin_amdgcn_wmma_f32_16x16x32_bf16(false, a.v, false, b.v,
                                                              (short)0, acc, false, false);
            }
            const float a_dt = (s < 2) ? cf.x : cf.z;
            const float b_dt = (s < 2) ? cf.y : cf.w;
            #pragma unroll
            for (int r = 0; r < 8; ++r) {
                float pre = acc[r] + bhh_c;
                float z = (pre - mu_c) * sg_c;
                float w = 1.f / (1.f + __expf(-z));
                h[r] = a_dt * h[r] + b_dt * (w * pre);
                __bf16 hv = (__bf16)h[r];
                int row = drow0 + r;
                nxt[(size_t)row * LTC_HID + col] = hv;
                if (s == 2) Hhist[((size_t)t * LTC_BATCH + row) * LTC_HID + col] = hv;
            }
            // ---- device-wide barrier (monotonic counter, reset each call by ltc_prep) ----
            __threadfence();
            __syncthreads();
            ++stepno;
            if (tid == 0) {
                __hip_atomic_fetch_add(bar, 1u, __ATOMIC_RELEASE, __HIP_MEMORY_SCOPE_AGENT);
                while (__hip_atomic_load(bar, __ATOMIC_ACQUIRE, __HIP_MEMORY_SCOPE_AGENT)
                       < (unsigned)LTC_NBLK * stepno)
                    __builtin_amdgcn_s_sleep(2);
            }
            __syncthreads();
            __threadfence();
            __bf16* tmp = cur; cur = nxt; nxt = tmp;   // swap double buffers
        }
    }
    // final hidden state (fp32) -> d_out tail
    #pragma unroll
    for (int r = 0; r < 8; ++r)
        hfinal[(size_t)(drow0 + r) * LTC_HID + col] = h[r];
}

// ---------------- batched output GEMM: outs[b,t,o] = Hhist @ W_out^T + b_out ----------------
// M = S*B = 32768 rows (row = t*128+b), N = 256, K = 1024. 16x64 tile per wave.
__global__ __launch_bounds__(256) void ltc_outgemm(
    const __bf16* __restrict__ Hhist, const __bf16* __restrict__ Woutb,
    const float* __restrict__ b_out, float* __restrict__ out)
{
    const int tid = threadIdx.x;
    const int wave = tid >> 5, lane = tid & 31;
    const int hi = lane >> 4, lq = lane & 15;
    const int job = blockIdx.x * 8 + wave;   // 8192 jobs = 2048 row tiles x 4 col groups
    const int rt = job >> 2, cg = job & 3;

    const __bf16* arow_p = Hhist + (size_t)(rt * 16 + lq) * LTC_HID;
    v8f acc[4] = {};
    #pragma unroll 2
    for (int kb = 0; kb < LTC_HID; kb += 32) {
        U16bf a;
        a.h[0] = *(const v8bf*)(arow_p + kb + hi * 8);
        a.h[1] = *(const v8bf*)(arow_p + kb + hi * 8 + 16);
        #pragma unroll
        for (int c = 0; c < 4; ++c) {
            int n = cg * 64 + c * 16 + lq;
            U16bf b;
            b.h[0] = *(const v8bf*)(Woutb + (size_t)n * LTC_HID + kb + hi * 8);
            b.h[1] = *(const v8bf*)(Woutb + (size_t)n * LTC_HID + kb + hi * 8 + 16);
            acc[c] = __builtin_amdgcn_wmma_f32_16x16x32_bf16(false, a.v, false, b.v,
                                                             (short)0, acc[c], false, false);
        }
    }
    #pragma unroll
    for (int c = 0; c < 4; ++c) {
        int o = cg * 64 + c * 16 + lq;
        float bo = b_out[o];
        #pragma unroll
        for (int r = 0; r < 8; ++r) {
            int row = rt * 16 + hi * 8 + r;
            int t = row >> 7, bidx = row & 127;
            out[((size_t)bidx * LTC_SEQ + t) * LTC_OUTW + o] = acc[c][r] + bo;
        }
    }
}

extern "C" void kernel_launch(void* const* d_in, const int* in_sizes, int n_in,
                              void* d_out, int out_size, void* d_ws, size_t ws_size,
                              hipStream_t stream) {
    const float* x       = (const float*)d_in[0];
    const float* ts      = (const float*)d_in[1];
    const float* W_in    = (const float*)d_in[2];
    const float* b_in    = (const float*)d_in[3];
    const float* W_hh    = (const float*)d_in[4];
    const float* b_hh    = (const float*)d_in[5];
    const float* log_tau = (const float*)d_in[6];
    const float* mu      = (const float*)d_in[7];
    const float* sigma   = (const float*)d_in[8];
    const float* gleak   = (const float*)d_in[9];
    const float* cm_t    = (const float*)d_in[10];
    const float* W_out   = (const float*)d_in[11];
    const float* b_out   = (const float*)d_in[12];
    float* out = (float*)d_out;

    // workspace carve-up (all chunks 256B-aligned by construction)
    char* p = (char*)d_ws;
    __bf16* Xb    = (__bf16*)p;  p += (size_t)LTC_BATCH * LTC_SEQ * LTC_IN * 2;   // 32 MB
    __bf16* Whhb  = (__bf16*)p;  p += (size_t)LTC_HID * LTC_HID * 2;              // 2 MB
    __bf16* Winb  = (__bf16*)p;  p += (size_t)LTC_HID * LTC_IN * 2;               // 1 MB
    __bf16* Woutb = (__bf16*)p;  p += (size_t)LTC_OUTW * LTC_HID * 2;             // 0.5 MB
    __bf16* hb0   = (__bf16*)p;  p += (size_t)LTC_BATCH * LTC_HID * 2;            // 256 KB
    __bf16* hb1   = (__bf16*)p;  p += (size_t)LTC_BATCH * LTC_HID * 2;            // 256 KB
    float4* coef  = (float4*)p;  p += (size_t)LTC_HID * sizeof(float4);           // 16 KB
    unsigned int* bar = (unsigned int*)p; p += 256;
    __bf16* Hhist = (__bf16*)p;  p += (size_t)LTC_SEQ * LTC_BATCH * LTC_HID * 2;  // 64 MB

    float* hfinal = out + (size_t)LTC_BATCH * LTC_SEQ * LTC_OUTW;

    ltc_cvt_bf16<<<2048, 256, 0, stream>>>(x, Xb, LTC_BATCH * LTC_SEQ * LTC_IN);
    ltc_cvt_bf16<<<1024, 256, 0, stream>>>(W_hh, Whhb, LTC_HID * LTC_HID);
    ltc_cvt_bf16<<<512,  256, 0, stream>>>(W_in, Winb, LTC_HID * LTC_IN);
    ltc_cvt_bf16<<<256,  256, 0, stream>>>(W_out, Woutb, LTC_OUTW * LTC_HID);
    ltc_prep<<<512, 256, 0, stream>>>(log_tau, gleak, cm_t, ts, coef, hb0, hb1, bar);

    ltc_persistent<<<LTC_NBLK, 256, 0, stream>>>(Xb, Winb, Whhb, hb0, hb1, Hhist,
                                                 b_in, b_hh, mu, sigma, coef, hfinal, bar);

    ltc_outgemm<<<1024, 256, 0, stream>>>(Hhist, Woutb, b_out, out);
}
